// Attention_86698209837214
// MI455X (gfx1250) — matrix operand order repaired and checked by reading, not compiled, not run on hardware
//
#include <hip/hip_runtime.h>
#include <hip/hip_bf16.h>

typedef __attribute__((ext_vector_type(16))) _Float16 v16h;
typedef __attribute__((ext_vector_type(8)))  _Float16 v8h;
typedef __attribute__((ext_vector_type(4)))  _Float16 v4h;
typedef __attribute__((ext_vector_type(8)))  float    v8f;
typedef __attribute__((ext_vector_type(4)))  int      v4i;

#define DIMX     1024
#define HEADS    16
#define DIM_HEAD 64
#define INNER    1024
#define SEQ      2048
#define BATCH    4
#define ROWS     (BATCH * SEQ)   // 8192
#define QKV3     (3 * INNER)     // 3072
#define PSTRIDE  40              // LDS P row stride (f16 elems; 80B, 16B-aligned)
#define BSTRIDE  40              // LDS GEMM B-tile row stride (32 data + 8 pad)
#define KSTRIDE  80              // LDS K-tile row stride  (64 data + 16 pad)
#define VSTRIDE  40              // LDS V-tile row stride  (32 data + 8 pad)

// ------------------------------------------------------- CDNA5 async copies
#if defined(__has_builtin)
#if __has_builtin(__builtin_amdgcn_global_load_async_to_lds_b128)
#define USE_ASYNC_BUILTIN 1
#endif
#if __has_builtin(__builtin_amdgcn_s_wait_asynccnt)
#define USE_WAIT_ASYNC_BUILTIN 1
#endif
#endif

// 16B global -> LDS async copy (ASYNCcnt-tracked on gfx1250)
__device__ __forceinline__ void async_copy_b128(_Float16* lds_dst,
                                                const _Float16* gsrc) {
#ifdef USE_ASYNC_BUILTIN
  // Toolchain declares the builtin with generic `v4i*` params
  // (diagnostic-confirmed); codegen inserts the addrspace casts.
  __builtin_amdgcn_global_load_async_to_lds_b128((v4i*)gsrc, (v4i*)lds_dst, 0,
                                                 0);
#else
  unsigned loff =
      (unsigned)(__SIZE_TYPE__)(__attribute__((address_space(3))) char*)(void*)
          lds_dst;
  asm volatile("global_load_async_to_lds_b128 %0, %1, off" ::"v"(loff),
               "v"(gsrc)
               : "memory");
#endif
}

__device__ __forceinline__ void wait_async0() {
#ifdef USE_WAIT_ASYNC_BUILTIN
  __builtin_amdgcn_s_wait_asynccnt(0);
#else
  asm volatile("s_wait_asynccnt 0x0" ::: "memory");
#endif
}

// ------------------------------------------------------- WMMA helpers
__device__ __forceinline__ v8f wmma16x16x32(v16h a, v16h b, v8f c) {
  // D = A(16x32 f16) x B(32x16 f16) + C(16x16 f32)
  return __builtin_amdgcn_wmma_f32_16x16x32_f16(false, a, false, b, (short)0, c,
                                                false, false);
}

__device__ __forceinline__ v16h cat8(v8h lo, v8h hi) {
  return __builtin_shufflevector(lo, hi, 0, 1, 2, 3, 4, 5, 6, 7, 8, 9, 10, 11,
                                 12, 13, 14, 15);
}

// A fragment (16x32 f16, row-major source). Contiguous k order:
//   lanes 0-15 : K[0..15]   (akb = 0)
//   lanes 16-31: K[16..31]  (akb = 8)
__device__ __forceinline__ v16h load_a_frag(const _Float16* row, int akb) {
  v8h lo = *(const v8h*)(row + 2 * akb);
  v8h hi = *(const v8h*)(row + 2 * akb + 8);
  return cat8(lo, hi);
}

// ---------------------------------------------------------------- conversions
__global__ void cvt_f32_to_f16_kernel(const float* __restrict__ in,
                                      _Float16* __restrict__ out, int n4) {
  int i = blockIdx.x * blockDim.x + threadIdx.x;
  int stride = gridDim.x * blockDim.x;
  for (; i < n4; i += stride) {
    float4 f = ((const float4*)in)[i];
    v4h h = {(_Float16)f.x, (_Float16)f.y, (_Float16)f.z, (_Float16)f.w};
    ((v4h*)out)[i] = h;
  }
}

// in [rows][cols] f32 -> out [cols][rows] f16 (B operands become col-major)
__global__ void transpose_cvt_kernel(const float* __restrict__ in,
                                     _Float16* __restrict__ out, int rows,
                                     int cols) {
  long idx = (long)blockIdx.x * blockDim.x + threadIdx.x;
  long total = (long)rows * cols;
  long stride = (long)gridDim.x * blockDim.x;
  for (; idx < total; idx += stride) {
    int r = (int)(idx / cols);
    int c = (int)(idx % cols);
    out[(long)c * rows + r] = (_Float16)in[idx];
  }
}

// ---------------------------------------------------------------- QKV GEMM
// xh[8192,1024] @ W_qkv -> q [B,H,N,Dh] (pre-scaled), k [B,H,N,Dh],
// vT [B,H,Dh,N].  wt = W_qkv^T as [3072,1024].
// Block = 4 waves, 64x64 tile; B tile (64x32) double-buffered in LDS via
// async copies, shared by all 4 waves.
__global__ void __launch_bounds__(128) gemm_qkv_kernel(
    const _Float16* __restrict__ xh, const _Float16* __restrict__ wt,
    _Float16* __restrict__ qbuf, _Float16* __restrict__ kbuf,
    _Float16* __restrict__ vtbuf) {
  __shared__ __align__(16) _Float16 sB[2][64 * BSTRIDE];

  const int tid = threadIdx.x;
  const int lane = tid & 31;
  const int wave = tid >> 5;
  const int ln = lane & 15;
  const int half = lane >> 4;
  const int akb = half * 8;
  const int bkb = half * 16;

  const int m0 = blockIdx.x * 64 + wave * 16;
  const int n0 = blockIdx.y * 64;

  const _Float16* arow = xh + (long)(m0 + ln) * DIMX;

  // prologue: stage k0 = 0  (256 x 16B segments, 2 per thread)
#pragma unroll
  for (int s = tid; s < 256; s += 128) {
    int row = s >> 2, seg = s & 3;
    async_copy_b128(&sB[0][row * BSTRIDE + seg * 8],
                    wt + (long)(n0 + row) * DIMX + seg * 8);
  }

  v8f acc0 = {}, acc1 = {}, acc2 = {}, acc3 = {};
  for (int k0 = 0; k0 < DIMX; k0 += 32) {
    const int cur = (k0 >> 5) & 1;
    wait_async0();    // my copies for sB[cur] landed
    __syncthreads();  // everyone's copies visible (also flushes DScnt)
    if (k0 + 32 < DIMX) {
#pragma unroll
      for (int s = tid; s < 256; s += 128) {
        int row = s >> 2, seg = s & 3;
        async_copy_b128(&sB[cur ^ 1][row * BSTRIDE + seg * 8],
                        wt + (long)(n0 + row) * DIMX + (k0 + 32) + seg * 8);
      }
    }
    v16h a = load_a_frag(arow + k0, akb);
    const _Float16* bb = &sB[cur][bkb];
    v16h b0 = *(const v16h*)(bb + (ln + 0) * BSTRIDE);
    v16h b1 = *(const v16h*)(bb + (ln + 16) * BSTRIDE);
    v16h b2 = *(const v16h*)(bb + (ln + 32) * BSTRIDE);
    v16h b3 = *(const v16h*)(bb + (ln + 48) * BSTRIDE);
    acc0 = wmma16x16x32(a, b0, acc0);
    acc1 = wmma16x16x32(a, b1, acc1);
    acc2 = wmma16x16x32(a, b2, acc2);
    acc3 = wmma16x16x32(a, b3, acc3);
  }

  const float scale = 0.125f;  // DIM_HEAD^-0.5, folded into q
  v8f accs[4] = {acc0, acc1, acc2, acc3};
#pragma unroll
  for (int t = 0; t < 4; ++t) {
    int gn = n0 + t * 16 + ln;
    int sec = gn >> 10;         // 0=q 1=k 2=v
    int hc = (gn & 1023) >> 6;  // head
    int d = gn & 63;            // dim within head
#pragma unroll
    for (int r = 0; r < 8; ++r) {
      int row = m0 + half * 8 + r;  // C/D layout: lanes>=16 hold M=r+8
      int bi = row >> 11;
      int ni = row & 2047;
      long bh = (long)(bi * HEADS + hc);
      float val = accs[t][r];
      if (sec == 0)
        qbuf[(bh * SEQ + ni) * DIM_HEAD + d] = (_Float16)(val * scale);
      else if (sec == 1)
        kbuf[(bh * SEQ + ni) * DIM_HEAD + d] = (_Float16)val;
      else
        vtbuf[(bh * DIM_HEAD + d) * SEQ + ni] = (_Float16)val;
    }
  }
}

// ---------------------------------------------------------------- attention
__device__ __forceinline__ float redmax16(float v) {
  v = fmaxf(v, __shfl_xor(v, 1, 32));
  v = fmaxf(v, __shfl_xor(v, 2, 32));
  v = fmaxf(v, __shfl_xor(v, 4, 32));
  v = fmaxf(v, __shfl_xor(v, 8, 32));
  return v;
}
__device__ __forceinline__ float redsum16(float v) {
  v += __shfl_xor(v, 1, 32);
  v += __shfl_xor(v, 2, 32);
  v += __shfl_xor(v, 4, 32);
  v += __shfl_xor(v, 8, 32);
  return v;
}

// stage K tile (32 x 64 halves) and V^T tile (64 x 32 halves) for keys
// [j0, j0+32) into LDS via async copies; 4 x 16B copies per thread.
__device__ __forceinline__ void stage_kv(_Float16* sk, _Float16* sv,
                                         const _Float16* kh,
                                         const _Float16* vh, int j0, int tid) {
#pragma unroll
  for (int s = tid; s < 256; s += 128) {  // K: 32 rows x 8 segs
    int row = s >> 3, seg = s & 7;
    async_copy_b128(sk + row * KSTRIDE + seg * 8,
                    kh + (long)(j0 + row) * DIM_HEAD + seg * 8);
  }
#pragma unroll
  for (int s = tid; s < 256; s += 128) {  // V^T: 64 rows x 4 segs
    int row = s >> 2, seg = s & 3;
    async_copy_b128(sv + row * VSTRIDE + seg * 8,
                    vh + (long)row * SEQ + j0 + seg * 8);
  }
}

// Flash attention: one wave per 16-query tile; 32 keys/iteration.
// All 4 waves of a block share the same (b,h), so K/V tiles are staged once
// per block in double-buffered LDS.
__global__ void __launch_bounds__(128) attn_kernel(
    const _Float16* __restrict__ q, const _Float16* __restrict__ k,
    const _Float16* __restrict__ vt, _Float16* __restrict__ ao) {
  __shared__ __align__(16) _Float16 sK[2][32 * KSTRIDE];
  __shared__ __align__(16) _Float16 sV[2][64 * VSTRIDE];
  __shared__ __align__(16) _Float16 s_p[4][16 * PSTRIDE];

  const int tid = threadIdx.x;
  const int lane = tid & 31;
  const int wave = tid >> 5;
  const int ln = lane & 15;
  const int half = lane >> 4;
  const int akb = half * 8;
  const int bkb = half * 16;

  const int qtile = blockIdx.x * 4 + wave;  // 4 qtiles/block, same head
  const int bh = qtile >> 7;                // b*HEADS + h (block-uniform)
  const int q0 = (qtile & 127) << 4;

  const _Float16* qh = q + (long)bh * SEQ * DIM_HEAD;
  const _Float16* kh = k + (long)bh * SEQ * DIM_HEAD;
  const _Float16* vh = vt + (long)bh * DIM_HEAD * SEQ;

  const _Float16* qrow = qh + (long)(q0 + ln) * DIM_HEAD;
  v16h qa0 = load_a_frag(qrow + 0, akb);   // Dh 0..31
  v16h qa1 = load_a_frag(qrow + 32, akb);  // Dh 32..63

  v8f o0 = {}, o1 = {}, o2 = {}, o3 = {};
  float mi[8], li[8];
#pragma unroll
  for (int r = 0; r < 8; ++r) {
    mi[r] = -3.0e38f;
    li[r] = 0.0f;
  }

  _Float16* sp = s_p[wave];

  stage_kv(sK[0], sV[0], kh, vh, 0, tid);  // prologue

  for (int j0 = 0; j0 < SEQ; j0 += 32) {
    const int cur = (j0 >> 5) & 1;
    wait_async0();
    __syncthreads();
    if (j0 + 32 < SEQ) stage_kv(sK[cur ^ 1], sV[cur ^ 1], kh, vh, j0 + 32, tid);

    // --- S = Q K^T for 2 key tiles (B operand = K rows from LDS)
    const _Float16* kr0 = &sK[cur][ln * KSTRIDE + bkb];
    const _Float16* kr1 = &sK[cur][(16 + ln) * KSTRIDE + bkb];
    v16h kb00 = *(const v16h*)(kr0);
    v16h kb01 = *(const v16h*)(kr0 + 32);
    v16h kb10 = *(const v16h*)(kr1);
    v16h kb11 = *(const v16h*)(kr1 + 32);

    v8f s0 = {}, s1 = {};
    s0 = wmma16x16x32(qa0, kb00, s0);
    s0 = wmma16x16x32(qa1, kb01, s0);
    s1 = wmma16x16x32(qa0, kb10, s1);
    s1 = wmma16x16x32(qa1, kb11, s1);

    // --- online softmax (row m = r + 8*half, key col = lane%16)
    float corr[8];
#pragma unroll
    for (int r = 0; r < 8; ++r) {
      float t = fmaxf(s0[r], s1[r]);
      t = redmax16(t);
      float mn = fmaxf(mi[r], t);
      corr[r] = __expf(mi[r] - mn);
      mi[r] = mn;
      float p0 = __expf(s0[r] - mn);
      float p1 = __expf(s1[r] - mn);
      li[r] = li[r] * corr[r] + redsum16(p0 + p1);
      int row = r + half * 8;
      sp[row * PSTRIDE + ln] = (_Float16)p0;
      sp[row * PSTRIDE + 16 + ln] = (_Float16)p1;
    }
#pragma unroll
    for (int r = 0; r < 8; ++r) {
      o0[r] *= corr[r];
      o1[r] *= corr[r];
      o2[r] *= corr[r];
      o3[r] *= corr[r];
    }
    __syncthreads();

    // --- reload P in A-fragment layout
    v8h plo = *(const v8h*)(sp + ln * PSTRIDE + 2 * akb);
    v8h phi = *(const v8h*)(sp + ln * PSTRIDE + 2 * akb + 8);
    v16h pa = cat8(plo, phi);

    // --- O += P V  (B operand = V^T rows from LDS)
    const _Float16* vb = &sV[cur][ln * VSTRIDE + bkb];
    v16h vb0 = *(const v16h*)(vb + 0 * 16 * VSTRIDE);
    v16h vb1 = *(const v16h*)(vb + 1 * 16 * VSTRIDE);
    v16h vb2 = *(const v16h*)(vb + 2 * 16 * VSTRIDE);
    v16h vb3 = *(const v16h*)(vb + 3 * 16 * VSTRIDE);
    o0 = wmma16x16x32(pa, vb0, o0);
    o1 = wmma16x16x32(pa, vb1, o1);
    o2 = wmma16x16x32(pa, vb2, o2);
    o3 = wmma16x16x32(pa, vb3, o3);
  }

  const int bi = bh >> 4;
  const int hh = bh & 15;
#pragma unroll
  for (int r = 0; r < 8; ++r) {
    float inv = 1.0f / li[r];
    int n = q0 + r + half * 8;
    _Float16* orow = ao + ((long)(bi * SEQ + n)) * INNER + hh * DIM_HEAD;
    orow[0 + ln] = (_Float16)(o0[r] * inv);
    orow[16 + ln] = (_Float16)(o1[r] * inv);
    orow[32 + ln] = (_Float16)(o2[r] * inv);
    orow[48 + ln] = (_Float16)(o3[r] * inv);
  }
}

// ---------------------------------------------------------------- out GEMM
// out[8192,1024] f32 = ao[8192,1024] @ W_out + b.  wot = W_out^T [1024,1024].
__global__ void __launch_bounds__(128) gemm_out_kernel(
    const _Float16* __restrict__ ao, const _Float16* __restrict__ wot,
    const float* __restrict__ bias, float* __restrict__ out) {
  __shared__ __align__(16) _Float16 sB[2][64 * BSTRIDE];

  const int tid = threadIdx.x;
  const int lane = tid & 31;
  const int wave = tid >> 5;
  const int ln = lane & 15;
  const int half = lane >> 4;
  const int akb = half * 8;
  const int bkb = half * 16;

  const int m0 = blockIdx.x * 64 + wave * 16;
  const int n0 = blockIdx.y * 64;

  const _Float16* arow = ao + (long)(m0 + ln) * INNER;

#pragma unroll
  for (int s = tid; s < 256; s += 128) {
    int row = s >> 2, seg = s & 3;
    async_copy_b128(&sB[0][row * BSTRIDE + seg * 8],
                    wot + (long)(n0 + row) * INNER + seg * 8);
  }

  v8f acc0 = {}, acc1 = {}, acc2 = {}, acc3 = {};
  for (int k0 = 0; k0 < INNER; k0 += 32) {
    const int cur = (k0 >> 5) & 1;
    wait_async0();
    __syncthreads();
    if (k0 + 32 < INNER) {
#pragma unroll
      for (int s = tid; s < 256; s += 128) {
        int row = s >> 2, seg = s & 3;
        async_copy_b128(&sB[cur ^ 1][row * BSTRIDE + seg * 8],
                        wot + (long)(n0 + row) * INNER + (k0 + 32) + seg * 8);
      }
    }
    v16h a = load_a_frag(arow + k0, akb);
    const _Float16* bb = &sB[cur][bkb];
    v16h b0 = *(const v16h*)(bb + (ln + 0) * BSTRIDE);
    v16h b1 = *(const v16h*)(bb + (ln + 16) * BSTRIDE);
    v16h b2 = *(const v16h*)(bb + (ln + 32) * BSTRIDE);
    v16h b3 = *(const v16h*)(bb + (ln + 48) * BSTRIDE);
    acc0 = wmma16x16x32(a, b0, acc0);
    acc1 = wmma16x16x32(a, b1, acc1);
    acc2 = wmma16x16x32(a, b2, acc2);
    acc3 = wmma16x16x32(a, b3, acc3);
  }

  v8f accs[4] = {acc0, acc1, acc2, acc3};
#pragma unroll
  for (int t = 0; t < 4; ++t) {
    int gn = n0 + t * 16 + ln;
    float b = bias[gn];
#pragma unroll
    for (int r = 0; r < 8; ++r) {
      int row = m0 + half * 8 + r;
      out[(long)row * DIMX + gn] = accs[t][r] + b;
    }
  }
}

// ---------------------------------------------------------------- launcher
extern "C" void kernel_launch(void* const* d_in, const int* in_sizes, int n_in,
                              void* d_out, int out_size, void* d_ws,
                              size_t ws_size, hipStream_t stream) {
  (void)in_sizes; (void)n_in; (void)out_size; (void)ws_size;
  const float* x = (const float*)d_in[0];     // [4,2048,1024]
  const float* Wqkv = (const float*)d_in[1];  // [1024,3072]
  const float* Wout = (const float*)d_in[2];  // [1024,1024]
  const float* bout = (const float*)d_in[3];  // [1024]

  char* ws = (char*)d_ws;
  const size_t MB = 1024 * 1024;
  _Float16* xh = (_Float16*)(ws + 0 * MB);    // 16 MB  x as f16
  _Float16* wqt = (_Float16*)(ws + 16 * MB);  //  6 MB  W_qkv^T f16
  _Float16* wot = (_Float16*)(ws + 22 * MB);  //  2 MB  W_out^T f16
  _Float16* qb = (_Float16*)(ws + 24 * MB);   // 16 MB  q [B,H,N,Dh] (scaled)
  _Float16* kb = (_Float16*)(ws + 40 * MB);   // 16 MB  k [B,H,N,Dh]
  _Float16* vtb = (_Float16*)(ws + 56 * MB);  // 16 MB  v^T [B,H,Dh,N]
  _Float16* aob = (_Float16*)(ws + 72 * MB);  // 16 MB  attn out [B,N,INNER]

  cvt_f32_to_f16_kernel<<<2048, 256, 0, stream>>>(x, xh, ROWS * DIMX / 4);
  transpose_cvt_kernel<<<2048, 256, 0, stream>>>(Wqkv, wqt, DIMX, QKV3);
  transpose_cvt_kernel<<<1024, 256, 0, stream>>>(Wout, wot, INNER, DIMX);

  dim3 g1(ROWS / 64, QKV3 / 64);
  gemm_qkv_kernel<<<g1, 128, 0, stream>>>(xh, wqt, qb, kb, vtb);

  int qtiles = BATCH * HEADS * (SEQ / 16);  // 8192
  attn_kernel<<<qtiles / 4, 128, 0, stream>>>(qb, kb, vtb, aob);

  dim3 g2(ROWS / 64, DIMX / 64);
  gemm_out_kernel<<<g2, 128, 0, stream>>>(aob, wot, bout, (float*)d_out);
}
